// BusStopPredictor_33921651703957
// MI455X (gfx1250) — matrix operand order repaired and checked by reading, not compiled, not run on hardware
//
#include <hip/hip_runtime.h>
#include <hip/hip_bf16.h>

typedef __attribute__((ext_vector_type(16))) _Float16 v16h;
typedef __attribute__((ext_vector_type(8)))  float    v8f;
typedef __attribute__((ext_vector_type(4)))  float    f32x4;

#define FDIM 64
#define NEG_SLOPE 0.2f
#define GAT_EPS 1e-16f

// ---------------------------------------------------------------------------
// Portable float atomic-max via sign-split int/uint atomics.
// Valid for any mix of signs when initialized to -inf (0xFF800000).
// ---------------------------------------------------------------------------
__device__ __forceinline__ void atomic_max_float(float* addr, float val) {
    if (val >= 0.0f) {
        atomicMax((int*)addr, __float_as_int(val));
    } else {
        atomicMin((unsigned int*)addr, __float_as_uint(val));
    }
}

// ---------------------------------------------------------------------------
// f32 -> f16 streaming convert
// ---------------------------------------------------------------------------
__global__ void k_f32_to_f16(const float* __restrict__ in, _Float16* __restrict__ out,
                             long n) {
    long i = (long)blockIdx.x * blockDim.x + threadIdx.x;
    if (i < n) out[i] = (_Float16)in[i];
}

// W [K=Fin][N=Fout] f32 row-major  ->  Wt [N][K] f16  (so WMMA B loads are contiguous)
__global__ void k_transpose_w16(const float* __restrict__ W, _Float16* __restrict__ Wt) {
    int i = blockIdx.x * blockDim.x + threadIdx.x;       // i = n*64 + k
    if (i < FDIM * FDIM) {
        int n = i >> 6, k = i & 63;
        Wt[i] = (_Float16)W[k * FDIM + n];
    }
}

// ---------------------------------------------------------------------------
// WMMA GEMM:  C[M,64] = A[M,64] (f16) x B[64,64] (f16, supplied as Bt[n][k])
// One wave per 16-row tile; 4 col tiles x 2 k-steps = 8 v_wmma per wave.
//
// f16 A 16x32 per-lane layout (ISA 7.12.2): lane<16 row=m0+l holds K {k0+0..7,
// k0+16..23}; lane>=16 same row group holds K {k0+8..15, k0+24..31}.
// f16 B 32x16: lane<16 col=n0+l holds K {k0+0..7, k0+16..23}; lane>=16 holds K {k0+8..15, k0+24..31}.
// f32 C/D 16x16: VGPR r, lane<16 -> (row m0+r, col n0+l); lane>=16 -> row m0+8+r.
// ---------------------------------------------------------------------------
__global__ void k_gemm_wmma_f16(const _Float16* __restrict__ A,
                                const _Float16* __restrict__ Bt,
                                float* __restrict__ C, int M) {
    const int lane = threadIdx.x & 31;
    const int wave = blockIdx.x * (blockDim.x >> 5) + (threadIdx.x >> 5);
    const int m0 = wave << 4;
    if (m0 >= M) return;                 // wave-uniform: EXEC stays all-ones
    const int half = lane >> 4;
    const int l    = lane & 15;

    v8f acc[4] = {};
    union U { v16h v; f32x4 q[2]; };

    #pragma unroll
    for (int k0 = 0; k0 < FDIM; k0 += 32) {
        U a;
        const _Float16* ap = A + (size_t)(m0 + l) * FDIM + (k0 + half * 8);
        a.q[0] = *(const f32x4*)ap;            // K k0+h*8 .. +7
        a.q[1] = *(const f32x4*)(ap + 16);     // K k0+16+h*8 .. +7
        #pragma unroll
        for (int nt = 0; nt < 4; ++nt) {
            U b;
            const _Float16* bp = Bt + (size_t)(nt * 16 + l) * FDIM + (k0 + half * 8);
            b.q[0] = *(const f32x4*)bp;        // K k0+h*8 .. +7
            b.q[1] = *(const f32x4*)(bp + 16); // K k0+16+h*8 .. +7
            acc[nt] = __builtin_amdgcn_wmma_f32_16x16x32_f16(
                false, a.v, false, b.v, (short)0, acc[nt], false, false);
        }
    }

    #pragma unroll
    for (int nt = 0; nt < 4; ++nt) {
        #pragma unroll
        for (int r = 0; r < 8; ++r) {
            C[(size_t)(m0 + half * 8 + r) * FDIM + nt * 16 + l] = acc[nt][r];
        }
    }
}

// ---------------------------------------------------------------------------
// Per-node attention scalars: asrc[n] = h[n,:]·a_src ; adst[n] = h[n,:]·a_dst
// One wave per node, 2 features per lane, shfl tree reduction (wave32).
// ---------------------------------------------------------------------------
__global__ void k_node_alpha(const float* __restrict__ h,
                             const float* __restrict__ a_src,
                             const float* __restrict__ a_dst,
                             float* __restrict__ asrc, float* __restrict__ adst, int N) {
    const int lane = threadIdx.x & 31;
    const int node = blockIdx.x * (blockDim.x >> 5) + (threadIdx.x >> 5);
    if (node >= N) return;
    const float* row = h + (size_t)node * FDIM;
    float x0 = row[lane], x1 = row[lane + 32];
    float s = x0 * a_src[lane] + x1 * a_src[lane + 32];
    float d = x0 * a_dst[lane] + x1 * a_dst[lane + 32];
    #pragma unroll
    for (int off = 16; off > 0; off >>= 1) {
        s += __shfl_down(s, off, 32);
        d += __shfl_down(d, off, 32);
    }
    if (lane == 0) { asrc[node] = s; adst[node] = d; }
}

// zero accumulator, init per-node softmax stats (-inf max, 0 denom)
__global__ void k_init(float* __restrict__ acc, float* __restrict__ emax,
                       float* __restrict__ denom, int N) {
    long i = (long)blockIdx.x * blockDim.x + threadIdx.x;
    long total = (long)N * FDIM;
    if (i < total) acc[i] = 0.0f;
    if (i < N) { emax[i] = -__builtin_inff(); denom[i] = 0.0f; }
}

// ---------------------------------------------------------------------------
// Edge passes.  i in [0, E+N): i<E is a real edge, i>=E is self-loop (i-E,i-E).
// ---------------------------------------------------------------------------
__device__ __forceinline__ float edge_score(int s, int d,
                                            const float* asrc, const float* adst) {
    float e = asrc[s] + adst[d];
    return e > 0.0f ? e : NEG_SLOPE * e;       // leaky_relu
}

__global__ void k_edge_max(const int* __restrict__ src, const int* __restrict__ dst,
                           const float* __restrict__ asrc, const float* __restrict__ adst,
                           float* __restrict__ emax, int E, int N) {
    int i = blockIdx.x * blockDim.x + threadIdx.x;
    if (i >= E + N) return;
    int s, d;
    if (i < E) { s = src[i]; d = dst[i]; } else { s = i - E; d = s; }
    atomic_max_float(&emax[d], edge_score(s, d, asrc, adst));
}

__global__ void k_edge_denom(const int* __restrict__ src, const int* __restrict__ dst,
                             const float* __restrict__ asrc, const float* __restrict__ adst,
                             const float* __restrict__ emax, float* __restrict__ denom,
                             int E, int N) {
    int i = blockIdx.x * blockDim.x + threadIdx.x;
    if (i >= E + N) return;
    int s, d;
    if (i < E) { s = src[i]; d = dst[i]; } else { s = i - E; d = s; }
    float ex = __expf(edge_score(s, d, asrc, adst) - emax[d]);
    atomicAdd(&denom[d], ex);
}

// Bandwidth-critical scatter: one wave per edge, 2 features/lane, coalesced
// global_load_b32 of h[src] + global_atomic_add_f32 into acc[dst].
__global__ void k_edge_aggregate(const int* __restrict__ src, const int* __restrict__ dst,
                                 const float* __restrict__ asrc, const float* __restrict__ adst,
                                 const float* __restrict__ emax, const float* __restrict__ denom,
                                 const float* __restrict__ h, float* __restrict__ acc,
                                 int E, int N) {
    const int lane = threadIdx.x & 31;
    const int i = blockIdx.x * (blockDim.x >> 5) + (threadIdx.x >> 5);
    if (i >= E + N) return;                    // wave-uniform
    int s, d;
    if (i < E) { s = src[i]; d = dst[i]; } else { s = i - E; d = s; }
    float alpha = __expf(edge_score(s, d, asrc, adst) - emax[d]) / (denom[d] + GAT_EPS);
    const float* hs = h + (size_t)s * FDIM;
    float* ad = acc + (size_t)d * FDIM;
    atomicAdd(&ad[lane],      alpha * hs[lane]);
    atomicAdd(&ad[lane + 32], alpha * hs[lane + 32]);
}

// acc = relu(acc + bias) in place; also emit f16 copy for next layer's WMMA input
__global__ void k_bias_relu(float* __restrict__ acc, const float* __restrict__ bias,
                            _Float16* __restrict__ out16, long total) {
    long i = (long)blockIdx.x * blockDim.x + threadIdx.x;
    if (i >= total) return;
    float v = acc[i] + bias[i & (FDIM - 1)];
    v = v > 0.0f ? v : 0.0f;
    acc[i] = v;
    out16[i] = (_Float16)v;
}

// ---------------------------------------------------------------------------
// Head: logits = h @ fcW + fcb (64 -> 2), then log_softmax. One wave per node.
// ---------------------------------------------------------------------------
__global__ void k_logits(const float* __restrict__ h, const float* __restrict__ fcW,
                         const float* __restrict__ fcb, float* __restrict__ out, int N) {
    const int lane = threadIdx.x & 31;
    const int node = blockIdx.x * (blockDim.x >> 5) + (threadIdx.x >> 5);
    if (node >= N) return;
    const float* row = h + (size_t)node * FDIM;
    float x0 = row[lane], x1 = row[lane + 32];
    float l0 = x0 * fcW[lane * 2 + 0] + x1 * fcW[(lane + 32) * 2 + 0];
    float l1 = x0 * fcW[lane * 2 + 1] + x1 * fcW[(lane + 32) * 2 + 1];
    #pragma unroll
    for (int off = 16; off > 0; off >>= 1) {
        l0 += __shfl_down(l0, off, 32);
        l1 += __shfl_down(l1, off, 32);
    }
    if (lane == 0) {
        l0 += fcb[0]; l1 += fcb[1];
        float m = fmaxf(l0, l1);
        float lse = m + __logf(__expf(l0 - m) + __expf(l1 - m));
        out[(size_t)node * 2 + 0] = l0 - lse;
        out[(size_t)node * 2 + 1] = l1 - lse;
    }
}

// ---------------------------------------------------------------------------
// Host side
// ---------------------------------------------------------------------------
static inline size_t align_up(size_t x, size_t a) { return (x + a - 1) & ~(a - 1); }

static void run_gat_layer(const _Float16* in16, const float* W, const float* a_src,
                          const float* a_dst, const float* bias,
                          const int* esrc, const int* edst,
                          _Float16* wt16, float* h, float* acc,
                          float* asrc, float* adst, float* emax, float* denom,
                          _Float16* out16, int N, int E, hipStream_t stream) {
    const int B = 256;
    const long total = (long)N * FDIM;
    const int ET = E + N;

    k_transpose_w16<<<(FDIM * FDIM + B - 1) / B, B, 0, stream>>>(W, wt16);

    int waves = (N + 15) / 16;
    k_gemm_wmma_f16<<<(waves + 7) / 8, B, 0, stream>>>(in16, wt16, h, N);

    k_node_alpha<<<(N + 7) / 8, B, 0, stream>>>(h, a_src, a_dst, asrc, adst, N);

    k_init<<<(int)((total + B - 1) / B), B, 0, stream>>>(acc, emax, denom, N);

    k_edge_max<<<(ET + B - 1) / B, B, 0, stream>>>(esrc, edst, asrc, adst, emax, E, N);
    k_edge_denom<<<(ET + B - 1) / B, B, 0, stream>>>(esrc, edst, asrc, adst, emax, denom, E, N);
    k_edge_aggregate<<<(ET + 7) / 8, B, 0, stream>>>(esrc, edst, asrc, adst, emax, denom,
                                                     h, acc, E, N);

    k_bias_relu<<<(int)((total + B - 1) / B), B, 0, stream>>>(acc, bias, out16, total);
}

extern "C" void kernel_launch(void* const* d_in, const int* in_sizes, int n_in,
                              void* d_out, int out_size, void* d_ws, size_t ws_size,
                              hipStream_t stream) {
    const float* x      = (const float*)d_in[0];
    const int*   eidx   = (const int*)  d_in[1];
    const float* W1     = (const float*)d_in[2];
    const float* a_src1 = (const float*)d_in[3];
    const float* a_dst1 = (const float*)d_in[4];
    const float* b1     = (const float*)d_in[5];
    const float* W2     = (const float*)d_in[6];
    const float* a_src2 = (const float*)d_in[7];
    const float* a_dst2 = (const float*)d_in[8];
    const float* b2     = (const float*)d_in[9];
    const float* fcW    = (const float*)d_in[10];
    const float* fcb    = (const float*)d_in[11];

    const int N = in_sizes[0] / FDIM;
    const int E = in_sizes[1] / 2;
    const int* esrc = eidx;
    const int* edst = eidx + E;

    // Workspace carving (~66 MB)
    const size_t nf = (size_t)N * FDIM;
    char* ws = (char*)d_ws;
    size_t off = 0;
    _Float16* hf16 = (_Float16*)(ws + off); off = align_up(off + nf * sizeof(_Float16), 256);
    _Float16* wt16 = (_Float16*)(ws + off); off = align_up(off + FDIM * FDIM * sizeof(_Float16), 256);
    float* h     = (float*)(ws + off); off = align_up(off + nf * sizeof(float), 256);
    float* acc   = (float*)(ws + off); off = align_up(off + nf * sizeof(float), 256);
    float* asrc  = (float*)(ws + off); off = align_up(off + (size_t)N * sizeof(float), 256);
    float* adst  = (float*)(ws + off); off = align_up(off + (size_t)N * sizeof(float), 256);
    float* emax  = (float*)(ws + off); off = align_up(off + (size_t)N * sizeof(float), 256);
    float* denom = (float*)(ws + off); off = align_up(off + (size_t)N * sizeof(float), 256);

    const int B = 256;
    const long total = (long)N * FDIM;

    // Layer 1: f16 input copy of x, then GAT layer -> acc (f32) + hf16 (f16)
    k_f32_to_f16<<<(int)((total + B - 1) / B), B, 0, stream>>>(x, hf16, total);
    run_gat_layer(hf16, W1, a_src1, a_dst1, b1, esrc, edst,
                  wt16, h, acc, asrc, adst, emax, denom, hf16, N, E, stream);

    // Layer 2 (hf16 now holds relu(layer1) as f16; acc is overwritten after GEMM)
    run_gat_layer(hf16, W2, a_src2, a_dst2, b2, esrc, edst,
                  wt16, h, acc, asrc, adst, emax, denom, hf16, N, E, stream);

    // Head: acc holds relu(layer2) in f32
    k_logits<<<(N + 7) / 8, B, 0, stream>>>(acc, fcW, fcb, (float*)d_out, N);
}
